// DeepSeekV2Attention_56650618635056
// MI455X (gfx1250) — matrix operand order repaired and checked by reading, not compiled, not run on hardware
//
#include <hip/hip_runtime.h>

typedef unsigned short u16;
typedef __attribute__((ext_vector_type(16))) __bf16 v16bf;
typedef __attribute__((ext_vector_type(8)))  float  v8f;
typedef __attribute__((ext_vector_type(4)))  unsigned int v4u;
typedef __attribute__((ext_vector_type(8)))  int v8i;
typedef __attribute__((ext_vector_type(4)))  int v4i;

#define B_    2
#define S_    2048
#define H_    2048
#define N_    16
#define NOPE_ 128
#define ROPE_ 64
#define V_    128
#define QHD_  192
#define QL_   1536
#define KVL_  512
#define SCALE_ 0.07216878364870322f   // 192^-0.5

static constexpr size_t BS_   = (size_t)B_ * S_;           // 4096 rows
static constexpr size_t OFF_QA   = 0;                      // fp32 4096x1536
static constexpr size_t SZ_QA    = BS_ * QL_ * 4;
static constexpr size_t OFF_ATTN = 0;                      // bf16 4096x2048 (aliases QA; QA dead by attention)
static constexpr size_t OFF_KVA  = OFF_QA + SZ_QA;         // fp32 4096x576
static constexpr size_t SZ_KVA   = BS_ * (KVL_ + ROPE_) * 4;
static constexpr size_t OFF_QAN  = OFF_KVA + SZ_KVA;       // bf16 4096x1536
static constexpr size_t SZ_QAN   = BS_ * QL_ * 2;
static constexpr size_t OFF_CKV  = OFF_QAN + SZ_QAN;       // bf16 4096x512
static constexpr size_t SZ_CKV   = BS_ * KVL_ * 2;
static constexpr size_t OFF_QB   = OFF_CKV + SZ_CKV;       // bf16 [B][N][S][192]
static constexpr size_t SZ_QB    = (size_t)B_ * N_ * S_ * QHD_ * 2;
static constexpr size_t OFF_KB   = OFF_QB + SZ_QB;         // bf16 [B][N][S][192]
static constexpr size_t SZ_KB    = SZ_QB;
static constexpr size_t OFF_VB   = OFF_KB + SZ_KB;         // bf16 V^T [B][N][128][S]
static constexpr size_t SZ_VB    = (size_t)B_ * N_ * S_ * V_ * 2;
static constexpr size_t OFF_BIG  = OFF_VB + SZ_VB;         // fp32 q (4096x3072) then kv (4096x4096), aliased

__device__ __forceinline__ u16 f2bf(float f) {
  unsigned u = __builtin_bit_cast(unsigned, f);
  u += 0x7fffu + ((u >> 16) & 1u);
  return (u16)(u >> 16);
}

__device__ __forceinline__ v8f zero8() {
  v8f z;
#pragma unroll
  for (int i = 0; i < 8; ++i) z[i] = 0.f;
  return z;
}

__device__ __forceinline__ unsigned lds_addr32(void* p) {
  return (unsigned)(unsigned long long)(uintptr_t)p;   // low 32 bits of generic ptr = LDS byte offset
}

// ---- Tensor Data Mover: 2-D bf16 tile (tile_d0 contiguous elems x tile_d1 rows) -> LDS ------
// pad_en/pad_int/pad_amt per D# group1 encoding (pad_int: 2^(n+1) DWORDs, pad_amt: n+1 DWORDs)
__device__ __forceinline__ void tdm_load_2d(unsigned lds, const void* gptr,
                                            unsigned tile_d0, unsigned tile_d1,
                                            unsigned long long stride_elems,
                                            unsigned pad_en, unsigned pad_int, unsigned pad_amt) {
  unsigned long long ga = (unsigned long long)(uintptr_t)gptr;
  v4u g0;
  g0[0] = 1u;                                            // count=1, user mode
  g0[1] = lds;                                           // lds_addr (bytes)
  g0[2] = (unsigned)ga;                                  // global_addr[31:0]
  g0[3] = (unsigned)((ga >> 32) & 0x01FFFFFFu) | 0x80000000u;  // addr[56:32] | type=2
  v8i g1;
  g1[0] = (int)((1u << 16) | (pad_en << 20) | (pad_int << 22) | (pad_amt << 25)); // data_size=2B
  g1[1] = (int)((tile_d0 & 0xFFFFu) << 16);              // tensor_dim0[15:0]
  g1[2] = (int)((tile_d0 >> 16) | ((tile_d1 & 0xFFFFu) << 16)); // td0 hi | tensor_dim1 lo
  g1[3] = (int)((tile_d1 >> 16) | (tile_d0 << 16));      // td1 hi | tile_dim0
  g1[4] = (int)(tile_d1 & 0xFFFFu);                      // tile_dim1 (tile_dim2=0)
  g1[5] = (int)(unsigned)(stride_elems & 0xFFFFFFFFu);   // tensor_dim0_stride lo
  g1[6] = (int)(unsigned)((stride_elems >> 32) & 0xFFFFu);
  g1[7] = 0;
  v4i gz; gz[0] = 0; gz[1] = 0; gz[2] = 0; gz[3] = 0;    // groups 2/3 unused (2-D)
  v8i gz8;
#pragma unroll
  for (int i = 0; i < 8; ++i) gz8[i] = 0;
  __builtin_amdgcn_tensor_load_to_lds(g0, g1, gz, gz, gz8, 0);
}

// A-fragment (16x32 bf16): lanes 0-15 -> K 0-15, lanes 16-31 -> K 16-31 (contiguous per lane)
__device__ __forceinline__ v16bf ld_afrag(const u16* rowp, int kb) {
  v16bf r;
#pragma unroll
  for (int e = 0; e < 8; ++e) {
    r[e]     = __builtin_bit_cast(__bf16, rowp[2 * kb + e]);
    r[e + 8] = __builtin_bit_cast(__bf16, rowp[2 * kb + 8 + e]);
  }
  return r;
}
// B-fragment (32x16 bf16): lanes 0-15 -> K 0-15, lanes 16-31 -> K 16-31 (contiguous per lane)
__device__ __forceinline__ v16bf ld_bfrag(const u16* rowp, int kb) {
  v16bf r;
#pragma unroll
  for (int e = 0; e < 16; ++e) r[e] = __builtin_bit_cast(__bf16, rowp[kb + e]);
  return r;
}

__device__ __forceinline__ void stage4(const float* p, u16* d) {
  const float4 v = *reinterpret_cast<const float4*>(p);
  d[0] = f2bf(v.x); d[1] = f2bf(v.y); d[2] = f2bf(v.z); d[3] = f2bf(v.w);
}

// ---------------- GEMM: C(MxN, f32) = A(MxK) * W(NxK)^T, bf16 WMMA, f32 acc -----------------
// Block tile 256x64, 8 waves, each wave 32x64 (8 WMMA / 32-K step).
// AT=u16 (bf16 A): A tile loaded by Tensor Data Mover with padded pitch 40.
template <typename AT>
__global__ __launch_bounds__(256) void gemm_wmma(const AT* __restrict__ A,
                                                 const float* __restrict__ W,
                                                 float* __restrict__ C,
                                                 int Nn, int K) {
  __shared__ u16 As[256][40];
  __shared__ u16 Ws[64][40];
  const int tid  = threadIdx.x;
  const int bm   = blockIdx.y * 256;
  const int bn   = blockIdx.x * 64;
  const int w    = tid >> 5, lane = tid & 31;
  const int wm   = w * 32;
  const int lm   = lane & 15, hb = lane >> 4;
  const int kbA  = hb ? 8 : 0, kbB = hb ? 16 : 0;

  v8f acc[2][4];
#pragma unroll
  for (int t = 0; t < 2; ++t)
#pragma unroll
    for (int u = 0; u < 4; ++u) acc[t][u] = zero8();

  for (int kt = 0; kt < K; kt += 32) {
#pragma unroll
    for (int i = 0; i < 2; ++i) {                 // W tile 64x32 f32 -> bf16
      int idx = tid + i * 256;
      int r = idx >> 3, c4 = (idx & 7) << 2;
      stage4(&W[(size_t)(bn + r) * K + kt + c4], &Ws[r][c4]);
    }
    if constexpr (sizeof(AT) == 2) {              // bf16 A: async TDM tile copy (wave 0 only)
      if (w == 0) {
        tdm_load_2d(lds_addr32(&As[0][0]), &A[(size_t)bm * K + kt],
                    32u, 256u, (unsigned long long)K,
                    1u, 3u, 3u);                  // pad every 16 DWORDs by 4 DWORDs -> pitch 40
        __builtin_amdgcn_s_wait_tensorcnt(0);
      }
    } else {                                      // f32 A: load + convert + stage
#pragma unroll
      for (int i = 0; i < 8; ++i) {
        int idx = tid + i * 256;                  // 0..2047 : A tile 256x32
        int r = idx >> 3, c4 = (idx & 7) << 2;
        stage4(&A[(size_t)(bm + r) * K + kt + c4], &As[r][c4]);
      }
      if (kt + 32 < K)
        __builtin_prefetch(&A[(size_t)(bm + (tid >> 1)) * K + kt + 32], 0, 0);
    }
    if (kt + 32 < K)
      __builtin_prefetch(&W[(size_t)(bn + (tid & 63)) * K + kt + 32], 0, 0);
    __syncthreads();

    v16bf af[2], bw[4];
#pragma unroll
    for (int t = 0; t < 2; ++t) af[t] = ld_afrag(&As[wm + t * 16 + lm][0], kbA);
#pragma unroll
    for (int u = 0; u < 4; ++u) bw[u] = ld_bfrag(&Ws[u * 16 + lm][0], kbB);
#pragma unroll
    for (int t = 0; t < 2; ++t)
#pragma unroll
      for (int u = 0; u < 4; ++u)
        acc[t][u] = __builtin_amdgcn_wmma_f32_16x16x32_bf16(
            false, af[t], false, bw[u], (short)0, acc[t][u], false, false);
    __syncthreads();
  }

#pragma unroll
  for (int t = 0; t < 2; ++t)
#pragma unroll
    for (int u = 0; u < 4; ++u)
#pragma unroll
      for (int r = 0; r < 8; ++r) {
        int m = bm + wm + t * 16 + (hb ? r + 8 : r);
        int c = bn + u * 16 + lm;
        C[(size_t)m * Nn + c] = acc[t][u][r];
      }
}

// ---------------- RMSNorm (row-per-block), f32 in -> bf16 out --------------------------------
__global__ __launch_bounds__(256) void rmsnorm_kernel(const float* __restrict__ in,
                                                      const float* __restrict__ w,
                                                      u16* __restrict__ out, int width) {
  __shared__ float red[256];
  const int row = blockIdx.x;
  const float* x = in + (size_t)row * width;
  float s = 0.f;
  for (int c = threadIdx.x; c < width; c += 256) { float v = x[c]; s += v * v; }
  red[threadIdx.x] = s; __syncthreads();
  for (int off = 128; off > 0; off >>= 1) {
    if (threadIdx.x < off) red[threadIdx.x] += red[threadIdx.x + off];
    __syncthreads();
  }
  const float inv = rsqrtf(red[0] / (float)width + 1e-6f);
  u16* o = out + (size_t)row * width;
  for (int c = threadIdx.x; c < width; c += 256) o[c] = f2bf(x[c] * inv * w[c]);
}

// ---------------- kv_a: rmsnorm(512) -> bf16, RoPE(k_pe 64) broadcast into K[...,128:192] ----
__global__ __launch_bounds__(128) void kv_norm_rope_kernel(const float* __restrict__ kva,
                                                           const float* __restrict__ w,
                                                           u16* __restrict__ ckv,
                                                           u16* __restrict__ Kb,
                                                           const int* __restrict__ pos) {
  __shared__ float red[128];
  const int row = blockIdx.x;                 // b*S + s
  const float* x = kva + (size_t)row * (KVL_ + ROPE_);
  float s = 0.f;
  for (int c = threadIdx.x; c < KVL_; c += 128) { float v = x[c]; s += v * v; }
  red[threadIdx.x] = s; __syncthreads();
  for (int off = 64; off > 0; off >>= 1) {
    if (threadIdx.x < off) red[threadIdx.x] += red[threadIdx.x + off];
    __syncthreads();
  }
  const float inv = rsqrtf(red[0] / (float)KVL_ + 1e-6f);
  for (int c = threadIdx.x; c < KVL_; c += 128)
    ckv[(size_t)row * KVL_ + c] = f2bf(x[c] * inv * w[c]);

  if (threadIdx.x < ROPE_) {
    const int j = threadIdx.x, jj = j & 31;
    const float p = (float)pos[row];
    const float ang = p * __powf(10000.f, -(float)(2 * jj) / 64.f);
    const float cv = __cosf(ang), sv = __sinf(ang);
    const float xq = x[KVL_ + j];
    const float other = (j < 32) ? -x[KVL_ + j + 32] : x[KVL_ + j - 32];
    const u16 val = f2bf(xq * cv + other * sv);
    const int b = row >> 11, ss = row & 2047;
#pragma unroll
    for (int n = 0; n < N_; ++n)
      Kb[(((size_t)(b * N_ + n) * S_) + ss) * QHD_ + NOPE_ + j] = val;
  }
}

// ---------------- q: RoPE on last 64, fold SCALE, scatter to Q[b][n][s][192] bf16 ------------
__global__ __launch_bounds__(256) void build_q_kernel(const float* __restrict__ q,
                                                      const int* __restrict__ pos,
                                                      u16* __restrict__ Qb) {
  const size_t idx = (size_t)blockIdx.x * 256 + threadIdx.x;   // 4096*3072
  const int row = (int)(idx / (N_ * QHD_));
  const int col = (int)(idx % (N_ * QHD_));
  const int n = col / QHD_, d = col % QHD_;
  float v;
  if (d < NOPE_) {
    v = q[idx];
  } else {
    const int j = d - NOPE_, jj = j & 31;
    const size_t base = (size_t)row * (N_ * QHD_) + n * QHD_ + NOPE_;
    const float p = (float)pos[row];
    const float ang = p * __powf(10000.f, -(float)(2 * jj) / 64.f);
    const float cv = __cosf(ang), sv = __sinf(ang);
    const float xq = q[base + j];
    const float other = (j < 32) ? -q[base + j + 32] : q[base + j - 32];
    v = xq * cv + other * sv;
  }
  const int b = row >> 11, ss = row & 2047;
  Qb[(((size_t)(b * N_ + n) * S_) + ss) * QHD_ + d] = f2bf(v * SCALE_);
}

// ---------------- kv: scatter to K[b][n][s][0:128] and V^T[b][n][vd][s] bf16 -----------------
__global__ __launch_bounds__(256) void build_kv_kernel(const float* __restrict__ kv,
                                                       u16* __restrict__ Kb,
                                                       u16* __restrict__ Vt) {
  const size_t idx = (size_t)blockIdx.x * 256 + threadIdx.x;   // 4096*4096
  const int row = (int)(idx >> 12);
  const int col = (int)(idx & 4095);
  const int n = col >> 8, d = col & 255;
  const float v = kv[idx];
  const int b = row >> 11, ss = row & 2047;
  if (d < NOPE_)
    Kb[(((size_t)(b * N_ + n) * S_) + ss) * QHD_ + d] = f2bf(v);
  else
    Vt[((size_t)(b * N_ + n) * V_ + (d - NOPE_)) * S_ + ss] = f2bf(v);
}

// ---------------- Flash attention: 1 wave per (b, head, 16-row q tile), 32-wide KV tiles -----
// Q/K/V tiles fetched by Tensor Data Mover (pure bf16 copies, V pre-transposed in global).
__global__ __launch_bounds__(32) void attn_kernel(const u16* __restrict__ Qb,
                                                  const u16* __restrict__ Kb,
                                                  const u16* __restrict__ Vt,
                                                  u16* __restrict__ Ob) {
  const int q0 = blockIdx.x * 16;
  const int n  = blockIdx.y;
  const int b  = blockIdx.z;
  const int lane = threadIdx.x, lm = lane & 15, hb = lane >> 4;
  const int kbA = hb ? 8 : 0, kbB = hb ? 16 : 0;

  __shared__ u16 Qs[16][192];
  __shared__ u16 Ks[32][192];
  __shared__ u16 Vs[128][32];   // V^T tile: [v-dim][kv-row]
  __shared__ u16 Ps[16][40];

  const size_t headQK = (size_t)(b * N_ + n) * S_;
  const size_t headVT = (size_t)(b * N_ + n) * V_ * S_;

  tdm_load_2d(lds_addr32(&Qs[0][0]), &Qb[(headQK + q0) * QHD_],
              (unsigned)QHD_, 16u, (unsigned long long)QHD_, 0u, 0u, 0u);
  __builtin_amdgcn_s_wait_tensorcnt(0);
  __syncthreads();

  v16bf qf[6];
#pragma unroll
  for (int c = 0; c < 6; ++c) qf[c] = ld_afrag(&Qs[lm][c * 32], kbA);

  v8f o[8];
#pragma unroll
  for (int t = 0; t < 8; ++t) o[t] = zero8();
  float mrun[8], lrun[8];
#pragma unroll
  for (int r = 0; r < 8; ++r) { mrun[r] = -3.0e38f; lrun[r] = 0.f; }

  const int ktiles = (q0 + 47) >> 5;
  for (int kt = 0; kt < ktiles; ++kt) {
    const int k0 = kt * 32;
    tdm_load_2d(lds_addr32(&Ks[0][0]), &Kb[(headQK + k0) * QHD_],
                (unsigned)QHD_, 32u, (unsigned long long)QHD_, 0u, 0u, 0u);
    tdm_load_2d(lds_addr32(&Vs[0][0]), &Vt[headVT + k0],
                32u, (unsigned)V_, (unsigned long long)S_, 0u, 0u, 0u);
    __builtin_amdgcn_s_wait_tensorcnt(0);
    __syncthreads();

    v8f s0 = zero8(), s1 = zero8();
#pragma unroll
    for (int c = 0; c < 6; ++c) {
      v16bf b0 = ld_bfrag(&Ks[lm][c * 32], kbB);
      v16bf b1 = ld_bfrag(&Ks[16 + lm][c * 32], kbB);
      s0 = __builtin_amdgcn_wmma_f32_16x16x32_bf16(false, qf[c], false, b0, (short)0, s0, false, false);
      s1 = __builtin_amdgcn_wmma_f32_16x16x32_bf16(false, qf[c], false, b1, (short)0, s1, false, false);
    }

    float alpha[8];
#pragma unroll
    for (int r = 0; r < 8; ++r) {
      const int rowid = hb ? r + 8 : r;
      const int rowg  = q0 + rowid;
      float x0 = (k0 + lm      <= rowg) ? s0[r] : -3.0e38f;
      float x1 = (k0 + 16 + lm <= rowg) ? s1[r] : -3.0e38f;
      float t = fmaxf(x0, x1);
#pragma unroll
      for (int m = 1; m < 16; m <<= 1) t = fmaxf(t, __shfl_xor(t, m, 32));
      const float mnew = fmaxf(mrun[r], t);
      const float a  = __expf(mrun[r] - mnew);
      const float p0 = __expf(x0 - mnew);
      const float p1 = __expf(x1 - mnew);
      float psum = p0 + p1;
#pragma unroll
      for (int m = 1; m < 16; m <<= 1) psum += __shfl_xor(psum, m, 32);
      lrun[r] = lrun[r] * a + psum;
      mrun[r] = mnew;
      alpha[r] = a;
      Ps[rowid][lm]      = f2bf(p0);
      Ps[rowid][16 + lm] = f2bf(p1);
    }
#pragma unroll
    for (int t = 0; t < 8; ++t)
#pragma unroll
      for (int r = 0; r < 8; ++r) o[t][r] *= alpha[r];

    __syncthreads();
    v16bf pf = ld_afrag(&Ps[lm][0], kbA);
#pragma unroll
    for (int t = 0; t < 8; ++t) {
      v16bf bv = ld_bfrag(&Vs[t * 16 + lm][0], kbB);
      o[t] = __builtin_amdgcn_wmma_f32_16x16x32_bf16(false, pf, false, bv, (short)0, o[t], false, false);
    }
    __syncthreads();
  }

#pragma unroll
  for (int r = 0; r < 8; ++r) {
    const int rowid = hb ? r + 8 : r;
    const float inv = 1.f / lrun[r];
    const size_t rowbase = ((size_t)(b * S_ + q0 + rowid)) * (N_ * V_) + n * V_;
#pragma unroll
    for (int t = 0; t < 8; ++t)
      Ob[rowbase + t * 16 + lm] = f2bf(o[t][r] * inv);
  }
}

extern "C" void kernel_launch(void* const* d_in, const int* in_sizes, int n_in,
                              void* d_out, int out_size, void* d_ws, size_t ws_size,
                              hipStream_t stream) {
  (void)in_sizes; (void)n_in; (void)out_size; (void)ws_size;
  const float* hs      = (const float*)d_in[0];
  const int*   pos     = (const int*)  d_in[1];
  const float* q_a_w   = (const float*)d_in[2];
  const float* q_a_ln  = (const float*)d_in[3];
  const float* q_b_w   = (const float*)d_in[4];
  const float* kv_a_w  = (const float*)d_in[5];
  const float* kv_a_ln = (const float*)d_in[6];
  const float* kv_b_w  = (const float*)d_in[7];
  const float* o_w     = (const float*)d_in[8];
  float* out = (float*)d_out;

  char* ws = (char*)d_ws;
  float* qa_f  = (float*)(ws + OFF_QA);
  float* kva_f = (float*)(ws + OFF_KVA);
  u16*   qan   = (u16*)(ws + OFF_QAN);
  u16*   ckv   = (u16*)(ws + OFF_CKV);
  u16*   Qb    = (u16*)(ws + OFF_QB);
  u16*   Kb    = (u16*)(ws + OFF_KB);
  u16*   Vt    = (u16*)(ws + OFF_VB);
  u16*   attnb = (u16*)(ws + OFF_ATTN);
  float* qbig  = (float*)(ws + OFF_BIG);   // 4096x3072 f32
  float* kvbig = (float*)(ws + OFF_BIG);   // 4096x4096 f32 (aliased; q consumed first)

  const dim3 blk(256);
  const int MB = (int)(BS_ / 256);         // 16

  gemm_wmma<float><<<dim3(QL_ / 64, MB), blk, 0, stream>>>(hs, q_a_w, qa_f, QL_, H_);
  gemm_wmma<float><<<dim3((KVL_ + ROPE_) / 64, MB), blk, 0, stream>>>(hs, kv_a_w, kva_f, KVL_ + ROPE_, H_);
  rmsnorm_kernel<<<(int)BS_, 256, 0, stream>>>(qa_f, q_a_ln, qan, QL_);
  kv_norm_rope_kernel<<<(int)BS_, 128, 0, stream>>>(kva_f, kv_a_ln, ckv, Kb, pos);
  gemm_wmma<u16><<<dim3((N_ * QHD_) / 64, MB), blk, 0, stream>>>(qan, q_b_w, qbig, N_ * QHD_, QL_);
  build_q_kernel<<<(int)(BS_ * N_ * QHD_ / 256), 256, 0, stream>>>(qbig, pos, Qb);
  gemm_wmma<u16><<<dim3((N_ * (NOPE_ + V_)) / 64, MB), blk, 0, stream>>>(ckv, kv_b_w, kvbig, N_ * (NOPE_ + V_), KVL_);
  build_kv_kernel<<<(int)(BS_ * N_ * (NOPE_ + V_) / 256), 256, 0, stream>>>(kvbig, Kb, Vt);
  attn_kernel<<<dim3(S_ / 16, N_, B_), 32, 0, stream>>>(Qb, Kb, Vt, attnb);
  gemm_wmma<u16><<<dim3(H_ / 64, MB), blk, 0, stream>>>(attnb, o_w, out, H_, N_ * V_);
}
